// Transformer_43877385896373
// MI455X (gfx1250) — matrix operand order repaired and checked by reading, not compiled, not run on hardware
//
#include <hip/hip_runtime.h>
#include <hip/hip_bf16.h>

typedef _Float16 half8_t  __attribute__((ext_vector_type(8)));
typedef _Float16 half16_t __attribute__((ext_vector_type(16)));
typedef float    float8_t __attribute__((ext_vector_type(8)));

#define LN_EPS 1e-5f

static constexpr int kBoards  = 64;   // T*B
static constexpr int kTok     = 81;   // H*W
static constexpr int kTokPad  = 96;   // 6 M-tiles of 16
static constexpr int kC       = 256;
static constexpr int kQKV     = 768;
static constexpr int kKSteps  = kC / 32;            // 8
static constexpr int kThreads = 512;
static constexpr int kWaves   = kThreads / 32;      // 16

// Packed-weight region sizes (halves)
static constexpr int kNT1     = kQKV / 16;          // 48 n-tiles, GEMM1 (3 per wave)
static constexpr int kNT2     = kC / 16;            // 16 n-tiles, GEMM2 (1 per wave)
static constexpr size_t kPk1Halves = (size_t)kNT1 * kKSteps * 32 * 16;   // 196,608
static constexpr size_t kPk2Halves = (size_t)kNT2 * kKSteps * 32 * 16;   //  65,536
static constexpr size_t kPkBytes   = (kPk1Halves + kPk2Halves) * sizeof(_Float16); // 524,288

union H16U { half16_t v; half8_t h[2]; _Float16 e[16]; };

// A fragment (16x32 f16) from LDS, row-major lda=kC.
// A 16x32 layout (matches B): lanes 0-15 row=lane, K 0..15;
// lanes 16-31 row=lane-16, K 16..31. Two contiguous 16B chunks per lane.
__device__ inline half16_t load_a_frag_lds(const _Float16* aBuf, int lane, int mtile, int kstep) {
    int row = mtile * 16 + (lane & 15);
    int k0  = kstep * 32 + ((lane >> 4) & 1) * 16;
    const _Float16* p = aBuf + row * kC + k0;
    H16U u;
    u.h[0] = *(const half8_t*)(p);
    u.h[1] = *(const half8_t*)(p + 8);
    return u.v;
}

// B fragment (32x16): lanes 0-15 hold K=0..15, lanes 16-31 K=16..31, N=lane%16.
template <bool PACKED>
__device__ inline half16_t load_b_frag(const float* W, const _Float16* Wp,
                                       int ldb, int lane, int ntile, int kstep) {
    H16U u;
    if constexpr (PACKED) {
        const _Float16* p = Wp + (((size_t)ntile * kKSteps + kstep) * 32 + lane) * 16;
        u.h[0] = *(const half8_t*)(p);
        u.h[1] = *(const half8_t*)(p + 8);
    } else {
        int col = ntile * 16 + (lane & 15);
        int k0  = kstep * 32 + ((lane >> 4) & 1) * 16;
        const float* p = W + (size_t)k0 * ldb + col;
#pragma unroll
        for (int i = 0; i < 16; ++i) u.e[i] = (_Float16)p[(size_t)i * ldb];
    }
    return u.v;
}

// qkv LDS layout per token row (768 floats): [ q(256) | k/v interleaved (512) ]
__device__ inline int qkv_store_addr(int m, int n) {
    if (n < kC)          return m * kQKV + n;                       // q
    else if (n < 2 * kC) return m * kQKV + kC + 2 * (n - kC);       // k
    else                 return m * kQKV + kC + 2 * (n - 2 * kC) + 1; // v
}

// One-time weight repack: f32 row-major -> f16 fragment-major (lane-contiguous 32B runs).
__global__ __launch_bounds__(256) void pack_weights_kernel(
    const float* __restrict__ w_qkv, const float* __restrict__ w_full, _Float16* __restrict__ ws)
{
    int t = blockIdx.x * blockDim.x + threadIdx.x;
    const int n1 = kNT1 * kKSteps * 32;   // 12288 lane-slots for GEMM1
    const int n2 = kNT2 * kKSteps * 32;   //  4096 lane-slots for GEMM2
    if (t < n1) {
        int lane = t & 31, f = t >> 5, ks = f % kKSteps, nt = f / kKSteps;
        int col = nt * 16 + (lane & 15);
        int k0  = ks * 32 + ((lane >> 4) & 1) * 16;
        _Float16* dst = ws + (size_t)t * 16;
#pragma unroll
        for (int i = 0; i < 16; ++i) dst[i] = (_Float16)w_qkv[(size_t)(k0 + i) * kQKV + col];
    } else if (t < n1 + n2) {
        int t2 = t - n1;
        int lane = t2 & 31, f = t2 >> 5, ks = f % kKSteps, nt = f / kKSteps;
        int col = nt * 16 + (lane & 15);
        int k0  = ks * 32 + ((lane >> 4) & 1) * 16;
        _Float16* dst = ws + kPk1Halves + (size_t)t2 * 16;
#pragma unroll
        for (int i = 0; i < 16; ++i) dst[i] = (_Float16)w_full[(size_t)(k0 + i) * kC + col];
    }
}

template <bool PACKED>
__global__ __launch_bounds__(kThreads) void board_attention_kernel(
    const float* __restrict__ x, const float* __restrict__ ln_g, const float* __restrict__ ln_b,
    const float* __restrict__ w_qkv, const float* __restrict__ b_qkv,
    const float* __restrict__ w_full, const float* __restrict__ b_full,
    const _Float16* __restrict__ wqkv_pk, const _Float16* __restrict__ wfull_pk,
    float* __restrict__ out)
{
    extern __shared__ char smem[];
    _Float16* aBuf   = (_Float16*)smem;                                   // kTokPad x kC (f16)
    float*    qkvBuf = (float*)(smem + kTokPad * kC * sizeof(_Float16));  // kTok x kQKV (f32)

    const int tid   = threadIdx.x;
    const int lane  = tid & 31;
    const int wave  = tid >> 5;
    const int board = blockIdx.x;
    const float* xb = x + (size_t)board * kTok * kC;

    // ---- Stage 0: zero pad rows (81..95) of A buffer ----
    for (int i = tid; i < (kTokPad - kTok) * kC; i += kThreads)
        aBuf[kTok * kC + i] = (_Float16)0.f;

    // ---- Stage 1: LayerNorm over C, one wave per token row ----
    for (int row = wave; row < kTok; row += kWaves) {
        const float* xr = xb + (size_t)row * kC;
        float v[8];
        float s = 0.f;
#pragma unroll
        for (int i = 0; i < 8; ++i) { v[i] = xr[lane + 32 * i]; s += v[i]; }
#pragma unroll
        for (int off = 16; off >= 1; off >>= 1) s += __shfl_xor(s, off, 32);
        float mu = s * (1.f / kC);
        float s2 = 0.f;
#pragma unroll
        for (int i = 0; i < 8; ++i) { float d = v[i] - mu; s2 += d * d; }
#pragma unroll
        for (int off = 16; off >= 1; off >>= 1) s2 += __shfl_xor(s2, off, 32);
        float rstd = rsqrtf(s2 * (1.f / kC) + LN_EPS);
#pragma unroll
        for (int i = 0; i < 8; ++i) {
            int c = lane + 32 * i;
            aBuf[row * kC + c] = (_Float16)((v[i] - mu) * rstd * ln_g[c] + ln_b[c]);
        }
    }
    __syncthreads();

    // ---- Stage 2: GEMM1  qkv = LN(x) @ w_qkv + b_qkv  (96x256 @ 256x768) ----
    // 3 n-tiles per wave, blocked: A fragments loaded once per k-step feed 18 WMMAs.
    {
        float8_t acc[3][6] = {};
        for (int ks = 0; ks < kKSteps; ++ks) {
            half16_t af[6];
#pragma unroll
            for (int mt = 0; mt < 6; ++mt) af[mt] = load_a_frag_lds(aBuf, lane, mt, ks);
            half16_t bf[3];
#pragma unroll
            for (int j = 0; j < 3; ++j)
                bf[j] = load_b_frag<PACKED>(w_qkv, wqkv_pk, kQKV, lane, wave + kWaves * j, ks);
#pragma unroll
            for (int j = 0; j < 3; ++j)
#pragma unroll
                for (int mt = 0; mt < 6; ++mt)
                    acc[j][mt] = __builtin_amdgcn_wmma_f32_16x16x32_f16(
                        false, af[mt], false, bf[j], (short)0, acc[j][mt], false, false);
        }
        int mo = ((lane >> 4) & 1) * 8;   // lanes>=16 hold M=8..15 of each tile
#pragma unroll
        for (int j = 0; j < 3; ++j) {
            int nt = wave + kWaves * j;
            int n  = nt * 16 + (lane & 15);
            float bias = b_qkv[n];
#pragma unroll
            for (int mt = 0; mt < 6; ++mt)
#pragma unroll
                for (int i = 0; i < 8; ++i) {
                    int m = mt * 16 + mo + i;
                    if (m < kTok) qkvBuf[qkv_store_addr(m, n)] = acc[j][mt][i] + bias;
                }
        }
    }
    __syncthreads();

    // ---- Stage 3: per-channel softmax attention (v_exp_f32-bound) ----
    {
        const int c = tid & (kC - 1);
        const int qphase = tid >> 8;           // 512 threads -> 2 q-phases per channel
        const float* kvp = qkvBuf + kC + 2 * c; // interleaved (k,v) pair for this channel
        float kmax = -1e30f, kmin = 1e30f;
        for (int r = 0; r < kTok; ++r) {
            float kv = kvp[r * kQKV];
            kmax = fmaxf(kmax, kv);
            kmin = fminf(kmin, kv);
        }
        const int jmax = (kTok - qphase + 1) >> 1;    // 41 or 40 q-values
        for (int jb = 0; jb < 42; jb += 7) {          // 7 q-values per pass
            float qv[7], mm[7], se[7], sev[7];
#pragma unroll
            for (int u = 0; u < 7; ++u) {
                int j = jb + u; if (j >= jmax) j = jmax - 1;   // clamp tail (redundant, race-free)
                int q = qphase + 2 * j;
                qv[u] = qkvBuf[q * kQKV + c];
                mm[u] = (qv[u] >= 0.f) ? qv[u] * kmax : qv[u] * kmin;  // exact row max
                se[u] = 0.f; sev[u] = 0.f;
            }
            for (int r = 0; r < kTok; ++r) {
                float2 kv = *(const float2*)(kvp + r * kQKV);  // one ds_load_b64 per row
#pragma unroll
                for (int u = 0; u < 7; ++u) {
                    float e = __expf(fmaf(qv[u], kv.x, -mm[u]));
                    se[u] += e;
                    sev[u] = fmaf(e, kv.y, sev[u]);
                }
            }
#pragma unroll
            for (int u = 0; u < 7; ++u) {
                int j = jb + u; if (j >= jmax) j = jmax - 1;
                int q = qphase + 2 * j;
                aBuf[q * kC + c] = (_Float16)(sev[u] / se[u]);  // reuse A buffer for GEMM2
            }
        }
    }
    __syncthreads();

    // ---- Stage 4: GEMM2  y = relu(att @ w_full + b_full); out = x + y ----
    for (int nt = wave; nt < kNT2; nt += kWaves) {
        float8_t acc[6] = {};
        for (int ks = 0; ks < kKSteps; ++ks) {
            half16_t bf = load_b_frag<PACKED>(w_full, wfull_pk, kC, lane, nt, ks);
            half16_t af[6];
#pragma unroll
            for (int mt = 0; mt < 6; ++mt) af[mt] = load_a_frag_lds(aBuf, lane, mt, ks);
#pragma unroll
            for (int mt = 0; mt < 6; ++mt)
                acc[mt] = __builtin_amdgcn_wmma_f32_16x16x32_f16(
                    false, af[mt], false, bf, (short)0, acc[mt], false, false);
        }
        int n = nt * 16 + (lane & 15);
        float bias = b_full[n];
        int mo = ((lane >> 4) & 1) * 8;
        float* ob = out + (size_t)board * kTok * kC;
#pragma unroll
        for (int mt = 0; mt < 6; ++mt)
#pragma unroll
            for (int i = 0; i < 8; ++i) {
                int m = mt * 16 + mo + i;
                if (m < kTok) {
                    float y = acc[mt][i] + bias;
                    y = y > 0.f ? y : 0.f;
                    size_t idx = (size_t)m * kC + n;
                    ob[idx] = xb[idx] + y;
                }
            }
    }
}

extern "C" void kernel_launch(void* const* d_in, const int* in_sizes, int n_in,
                              void* d_out, int out_size, void* d_ws, size_t ws_size,
                              hipStream_t stream) {
    const float* x      = (const float*)d_in[0];
    const float* ln_g   = (const float*)d_in[1];
    const float* ln_b   = (const float*)d_in[2];
    const float* w_qkv  = (const float*)d_in[3];
    const float* b_qkv  = (const float*)d_in[4];
    const float* w_full = (const float*)d_in[5];
    const float* b_full = (const float*)d_in[6];
    float* out = (float*)d_out;

    size_t lds_bytes = (size_t)kTokPad * kC * sizeof(_Float16)   // 49,152 B  A operand
                     + (size_t)kTok * kQKV * sizeof(float);      // 248,832 B qkv staging

    const bool packed = (d_ws != nullptr) && (ws_size >= kPkBytes);
    if (packed) {
        _Float16* wsp = (_Float16*)d_ws;
        int slots = (kNT1 + kNT2) * kKSteps * 32;               // 16,384 lane-slots
        pack_weights_kernel<<<dim3((slots + 255) / 256), dim3(256), 0, stream>>>(w_qkv, w_full, wsp);
        board_attention_kernel<true><<<dim3(kBoards), dim3(kThreads), lds_bytes, stream>>>(
            x, ln_g, ln_b, w_qkv, b_qkv, w_full, b_full,
            wsp, wsp + kPk1Halves, out);
    } else {
        board_attention_kernel<false><<<dim3(kBoards), dim3(kThreads), lds_bytes, stream>>>(
            x, ln_g, ln_b, w_qkv, b_qkv, w_full, b_full,
            nullptr, nullptr, out);
    }
}
